// QuantizedMatryoshkaDecoder_75651553951803
// MI455X (gfx1250) — matrix operand order repaired and checked by reading, not compiled, not run on hardware
//
#include <hip/hip_runtime.h>
#include <cstdint>
#include <cstddef>

// ---------------- problem geometry ----------------
constexpr int MDIM = 4096;    // batch
constexpr int NDIM = 1024;    // out features
constexpr int KDIM = 16384;   // in features

constexpr int BM = 128;       // block tile M
constexpr int BN = 128;       // block tile N  (wider: amortizes A conversion 2x better)
constexpr int BK = 32;        // K step (matches wmma 16x16x32)
constexpr int KSTEPS = KDIM / BK;     // 512

// LDS strides in ushort units (padded vs bank conflicts, 16B aligned rows/cols)
constexpr int A_STRIDE = 40;  // 32 bf16 data + 8 pad = 80 bytes/row
constexpr int B_STRIDE = 48;  // 32 bf16 data + 16 pad = 96 bytes/col

typedef __attribute__((ext_vector_type(16))) __bf16 v16bf;
typedef __attribute__((ext_vector_type(8)))  float  v8f;
typedef __attribute__((ext_vector_type(4)))  float  f32x4;
typedef __attribute__((ext_vector_type(4)))  unsigned int u32x4;
typedef __attribute__((ext_vector_type(2)))  unsigned int u32x2;

union Frag16 { u32x4 q[2]; v16bf v; };

// RNE-rounded f32->bf16 as integer bits (rounded value reusable for hi-as-f32)
__device__ __forceinline__ unsigned bf16_round_bits(float f) {
  unsigned u = __float_as_uint(f);
  return u + 0x7FFFu + ((u >> 16) & 1u);
}

__device__ __forceinline__ float slice_scale(int k0) {
  // slice i covers [128<<(i-1), 128<<i), slice 0 = [0,128)
  const int s = (k0 < 128) ? 0 : (32 - __builtin_clz((unsigned)(k0 >> 7)));
  return (4.0f / 255.0f) * (float)(1 << (7 - s));   // 2^(7-i) * QUANT_STEP
}

__global__ void __launch_bounds__(256)
matryoshka_wmma_kernel(const float* __restrict__ latent,
                       const float* __restrict__ weight,
                       float* __restrict__ out) {
  __shared__ __align__(16) unsigned short sAhi[2][BM * A_STRIDE];  // 2 x 10 KB
  __shared__ __align__(16) unsigned short sAlo[2][BM * A_STRIDE];  // 2 x 10 KB
  __shared__ __align__(16) unsigned short sB  [2][BN * B_STRIDE];  // 2 x 12 KB

  const int tid  = threadIdx.x;
  const int lane = tid & 31;
  const int wave = tid >> 5;
  const int wm   = wave >> 1;      // 0..3  -> 32-row strip
  const int wn   = wave & 1;       // 0..1  -> 64-col strip
  const int hlf  = lane >> 4;      // lane half (WMMA layout)
  const int l16  = lane & 15;

  const int bm0 = (int)blockIdx.x * BM;
  const int bn0 = (int)blockIdx.y * BN;

  // A staging: thread -> (row ar, 16 consecutive floats starting at ac)
  const int ar = tid >> 1;
  const int ac = (tid & 1) * 16;
  // B staging: thread -> (col bcol, 16 consecutive K rows starting at brow)
  const int bcol = tid & 127;
  const int brow = (tid >> 7) * 16;

  f32x4 aReg[4];
  float bReg[16];

  auto load_tile = [&](int kt) {
    const float* Ap = latent + (size_t)(bm0 + ar) * KDIM + (size_t)kt * BK + ac;
#pragma unroll
    for (int j = 0; j < 4; ++j) aReg[j] = *(const f32x4*)(Ap + j * 4);
    const float* Wp = weight + (size_t)(kt * BK + brow) * NDIM + (bn0 + bcol);
#pragma unroll
    for (int j = 0; j < 16; ++j) bReg[j] = Wp[(size_t)j * NDIM];
  };

  auto stage_tile = [&](int buf, float scale) {
    // A: scale latent, split into bf16 hi + lo residual (≈17-bit mantissa pair)
#pragma unroll
    for (int j = 0; j < 4; ++j) {
      float s[4] = { aReg[j].x * scale, aReg[j].y * scale,
                     aReg[j].z * scale, aReg[j].w * scale };
      unsigned r[4], lo[4];
#pragma unroll
      for (int e = 0; e < 4; ++e) {
        r[e] = bf16_round_bits(s[e]);
        const float hi_f = __uint_as_float(r[e] & 0xFFFF0000u);
        lo[e] = bf16_round_bits(s[e] - hi_f);
      }
      u32x2 hv, lv;
      hv.x = (r[0] >> 16) | (r[1] & 0xFFFF0000u);
      hv.y = (r[2] >> 16) | (r[3] & 0xFFFF0000u);
      lv.x = (lo[0] >> 16) | (lo[1] & 0xFFFF0000u);
      lv.y = (lo[2] >> 16) | (lo[3] & 0xFFFF0000u);
      const int off = ar * A_STRIDE + ac + j * 4;
      *(u32x2*)&sAhi[buf][off] = hv;
      *(u32x2*)&sAlo[buf][off] = lv;
    }
    // B: sign(weight) in bf16 (+1 = 0x3F80, -1 = 0xBF80), stored column-major
    unsigned p[8];
#pragma unroll
    for (int j = 0; j < 8; ++j) {
      const unsigned s0 = (bReg[2 * j]     >= 0.f) ? 0x3F80u : 0xBF80u;
      const unsigned s1 = (bReg[2 * j + 1] >= 0.f) ? 0x3F80u : 0xBF80u;
      p[j] = s0 | (s1 << 16);
    }
    u32x4 sv0, sv1;
    sv0.x = p[0]; sv0.y = p[1]; sv0.z = p[2]; sv0.w = p[3];
    sv1.x = p[4]; sv1.y = p[5]; sv1.z = p[6]; sv1.w = p[7];
    const int boff = bcol * B_STRIDE + brow;
    *(u32x4*)&sB[buf][boff]     = sv0;
    *(u32x4*)&sB[buf][boff + 8] = sv1;
  };

  // C accumulators: directly hold the running "recon" (scale folded into A)
  v8f acc[2][4];
#pragma unroll
  for (int mi = 0; mi < 2; ++mi)
#pragma unroll
    for (int ni = 0; ni < 4; ++ni)
#pragma unroll
      for (int e = 0; e < 8; ++e) acc[mi][ni][e] = 0.0f;

  load_tile(0);
  stage_tile(0, slice_scale(0));
  __syncthreads();

  int buf = 0;
  for (int kt = 0; kt < KSTEPS; ++kt) {
    const bool more = (kt + 1) < KSTEPS;
    if (more) load_tile(kt + 1);    // global prefetch of next tile into regs

    // ---- fragment loads (match ISA 16-bit A/B VGPR layouts) ----
    Frag16 ahi[2], alo[2], bfr[4];
#pragma unroll
    for (int mi = 0; mi < 2; ++mi) {
      const int row  = wm * 32 + mi * 16 + l16;
      const int base = row * A_STRIDE + hlf * 8;   // lane<16: K0..7/K16..23; lane>=16: K8..15/K24..31
      ahi[mi].q[0] = *(const u32x4*)&sAhi[buf][base];
      ahi[mi].q[1] = *(const u32x4*)&sAhi[buf][base + 16];
      alo[mi].q[0] = *(const u32x4*)&sAlo[buf][base];
      alo[mi].q[1] = *(const u32x4*)&sAlo[buf][base + 16];
    }
#pragma unroll
    for (int ni = 0; ni < 4; ++ni) {
      const int col  = wn * 64 + ni * 16 + l16;
      const int base = col * B_STRIDE + hlf * 8;   // lane<16: K0..7/K16..23; lane>=16: K8..15/K24..31
      bfr[ni].q[0] = *(const u32x4*)&sB[buf][base];
      bfr[ni].q[1] = *(const u32x4*)&sB[buf][base + 16];
    }

    // ---- 16 WMMAs per wave per K-step (hi + lo contributions) ----
#pragma unroll
    for (int mi = 0; mi < 2; ++mi)
#pragma unroll
      for (int ni = 0; ni < 4; ++ni) {
        acc[mi][ni] = __builtin_amdgcn_wmma_f32_16x16x32_bf16(
            false, ahi[mi].v, false, bfr[ni].v, (short)0, acc[mi][ni], false, false);
        acc[mi][ni] = __builtin_amdgcn_wmma_f32_16x16x32_bf16(
            false, alo[mi].v, false, bfr[ni].v, (short)0, acc[mi][ni], false, false);
      }

    if (more) stage_tile(buf ^ 1, slice_scale((kt + 1) * BK));
    __syncthreads();

    // ---- snapshot store at slice boundaries (kend = 128,256,...,16384) ----
    const int kend = (kt + 1) * BK;
    if (kend >= 128 && (kend & (kend - 1)) == 0) {
      const int stage = 31 - __builtin_clz((unsigned)(kend >> 7));   // 0..7
      float* outS = out + (size_t)stage * MDIM * NDIM;
#pragma unroll
      for (int mi = 0; mi < 2; ++mi)
#pragma unroll
        for (int ni = 0; ni < 4; ++ni) {
          const int row0 = bm0 + wm * 32 + mi * 16 + hlf * 8;  // C layout: vgpr r -> M=row0+r
          const int col  = bn0 + wn * 64 + ni * 16 + l16;      // N = lane
          float* p = outS + (size_t)row0 * NDIM + col;
#pragma unroll
          for (int r = 0; r < 8; ++r) p[(size_t)r * NDIM] = acc[mi][ni][r];
        }
    }
    buf ^= 1;
  }
}

extern "C" void kernel_launch(void* const* d_in, const int* in_sizes, int n_in,
                              void* d_out, int out_size, void* d_ws, size_t ws_size,
                              hipStream_t stream) {
  (void)in_sizes; (void)n_in; (void)d_ws; (void)ws_size; (void)out_size;
  const float* latent = (const float*)d_in[0];   // (4096, 16384) f32
  const float* weight = (const float*)d_in[1];   // (16384, 1024) f32
  float* out = (float*)d_out;                    // (8, 4096, 1024) f32

  dim3 grid(MDIM / BM, NDIM / BN);               // (32, 8) = 256 workgroups
  dim3 block(256);                               // 8 wave32s
  hipLaunchKernelGGL(matryoshka_wmma_kernel, grid, block, 0, stream,
                     latent, weight, out);
}
